// LocalFmoeCatEmbedFeedForward_86981677678859
// MI455X (gfx1250) — matrix operand order repaired and checked by reading, not compiled, not run on hardware
//
#include <hip/hip_runtime.h>
#include <hip/hip_bf16.h>

typedef __bf16 bf16_t;
typedef bf16_t v16bf __attribute__((ext_vector_type(16)));
typedef bf16_t v8bf  __attribute__((ext_vector_type(8)));
typedef float  v8f   __attribute__((ext_vector_type(8)));

#define NTOK   16384   // B*T
#define DID    512     // IDIM
#define EDIM   256
#define CDIM   768     // EDIM + DID
#define NEXP   4
#define HID    1024
#define TM     64      // tokens per FFN block (four 16-row WMMA tiles)
#define LDX    520     // LDS stride, x tile (bf16): 16B-aligned rows, conflict-free
#define LDH    1032    // LDS stride, h tile (bf16): 16B-aligned rows, conflict-free

// ---------------- workspace layout (bytes) ----------------
#define WS_GATE  256
#define WS_PERM  (WS_GATE + NTOK*4)
#define WS_W1    (WS_PERM + NEXP*NTOK*4)      // 16B aligned
#define WS_W2    (WS_W1 + NEXP*HID*DID*2)     // 16B aligned

#define WMMA_BF16(A,B,C) __builtin_amdgcn_wmma_f32_16x16x32_bf16( \
    false, (A), false, (B), (short)0, (C), false, false)

// ---- A fragment: 16x32 bf16, row-major src [16 x K], per ISA 7.12.2 ----
__device__ __forceinline__ v16bf ld_a(const bf16_t* base, int ld, int lane) {
    int r  = lane & 15;
    int kb = (lane >> 4) << 3;          // 0 or 8
    const bf16_t* p = base + r * ld + kb;
    v8bf lo = *(const v8bf*)p;          // K = kb .. kb+7
    v8bf hi = *(const v8bf*)(p + 16);   // K = kb+16 .. kb+23
    v16bf f;
#pragma unroll
    for (int i = 0; i < 8; ++i) { f[i] = lo[i]; f[i + 8] = hi[i]; }
    return f;
}

// ---- B fragment: 32x16 bf16 loaded from B^T row-major [16(N) x K] ----
__device__ __forceinline__ v16bf ld_b(const bf16_t* base, int ld, int lane) {
    int r  = lane & 15;                 // output column n
    int kb = (lane >> 4) << 3;          // 0 or 8
    const bf16_t* p = base + r * ld + kb;
    v8bf lo = *(const v8bf*)p;          // K = kb .. kb+7
    v8bf hi = *(const v8bf*)(p + 16);   // K = kb+16 .. kb+23
    v16bf f;
#pragma unroll
    for (int i = 0; i < 8; ++i) { f[i] = lo[i]; f[i + 8] = hi[i]; }
    return f;
}

// ================= kernel 0: convert weights to bf16, zero counters ======
__global__ void moe_prep(const float* __restrict__ w1, const float* __restrict__ w2,
                         bf16_t* __restrict__ w1b, bf16_t* __restrict__ w2b,
                         int* __restrict__ counts) {
    int gid = blockIdx.x * blockDim.x + threadIdx.x;
    if (gid < NEXP) counts[gid] = 0;
    const int n = NEXP * HID * DID;
    int stride = gridDim.x * blockDim.x;
    for (int i = gid; i < n; i += stride) {
        w1b[i] = (bf16_t)w1[i];
        w2b[i] = (bf16_t)w2[i];
    }
}

// ================= kernel 1: router (one wave32 per token) ===============
__global__ void moe_router(const float* __restrict__ inputs, const float* __restrict__ embed,
                           const float* __restrict__ rw, const unsigned char* __restrict__ mask,
                           float* __restrict__ gate, int* __restrict__ perm,
                           int* __restrict__ counts) {
    int wid  = threadIdx.x >> 5;
    int lane = threadIdx.x & 31;
    int token = blockIdx.x * 8 + wid;

    const float* erow = embed  + (size_t)token * EDIM;
    const float* xrow = inputs + (size_t)token * DID;
    float a0 = 0.f, a1 = 0.f, a2 = 0.f, a3 = 0.f;
    for (int c = lane; c < CDIM; c += 32) {
        float v = (c < EDIM) ? erow[c] : xrow[c - EDIM];
        float4 wv = *(const float4*)(rw + c * 4);
        a0 += v * wv.x; a1 += v * wv.y; a2 += v * wv.z; a3 += v * wv.w;
    }
#pragma unroll
    for (int off = 16; off > 0; off >>= 1) {
        a0 += __shfl_xor(a0, off, 32);
        a1 += __shfl_xor(a1, off, 32);
        a2 += __shfl_xor(a2, off, 32);
        a3 += __shfl_xor(a3, off, 32);
    }
    if (lane == 0) {
        float l[4] = {a0, a1, a2, a3};
        int ei = 0; float m = l[0];
#pragma unroll
        for (int e = 1; e < NEXP; ++e) if (l[e] > m) { m = l[e]; ei = e; }
        float s = __expf(l[0] - m) + __expf(l[1] - m) + __expf(l[2] - m) + __expf(l[3] - m);
        float gv = (mask[token] ? 1.0f : 0.0f) / s;   // top-1 prob = exp(0)/sum
        gate[token] = gv;
        int pos = atomicAdd(counts + ei, 1);
        perm[ei * NTOK + pos] = token;
    }
}

// ================= kernel 2: grouped FFN (64-token tile per block) =======
__global__ __launch_bounds__(256, 1)
void moe_ffn(const float* __restrict__ inputs,
             const float* __restrict__ b1, const float* __restrict__ b2,
             const bf16_t* __restrict__ w1b, const bf16_t* __restrict__ w2b,
             const float* __restrict__ gate, const int* __restrict__ perm,
             const int* __restrict__ counts, float* __restrict__ out) {
    __shared__ bf16_t sx[TM * LDX];     // gathered x tile   (64 x 512 bf16)  ~65 KB
    __shared__ bf16_t sh[TM * LDH];     // hidden tile       (64 x 1024 bf16) ~129 KB
    __shared__ int    s_tok[TM];
    __shared__ float  s_gate[TM];

    // ---- map blockIdx -> (expert, local tile) from routing counters ----
    int c0 = counts[0], c1 = counts[1], c2 = counts[2], c3 = counts[3];
    int t0 = (c0 + TM - 1) / TM, t1 = (c1 + TM - 1) / TM;
    int t2 = (c2 + TM - 1) / TM, t3 = (c3 + TM - 1) / TM;
    int bid = blockIdx.x;
    int e, lt, cnt;
    if      (bid < t0)                { e = 0; lt = bid;                 cnt = c0; }
    else if (bid < t0 + t1)           { e = 1; lt = bid - t0;            cnt = c1; }
    else if (bid < t0 + t1 + t2)      { e = 2; lt = bid - t0 - t1;       cnt = c2; }
    else if (bid < t0 + t1 + t2 + t3) { e = 3; lt = bid - t0 - t1 - t2;  cnt = c3; }
    else return;                        // uniform early-exit, before barriers

    int tid = threadIdx.x;
    if (tid < TM) {
        int p = lt * TM + tid;
        if (p < cnt) { int g = perm[e * NTOK + p]; s_tok[tid] = g; s_gate[tid] = gate[g]; }
        else         { s_tok[tid] = -1;            s_gate[tid] = 0.f; }
    }
    __syncthreads();

    // ---- gather + fp32->bf16 the 64 token rows into LDS (float4 loads) ----
    for (int i = tid * 4; i < TM * DID; i += 256 * 4) {
        int r = i >> 9, c = i & (DID - 1);
        int g = s_tok[r];
        bf16_t* dst = sx + r * LDX + c;
        if (g >= 0) {
            float4 v = *(const float4*)(inputs + (size_t)g * DID + c);
            dst[0] = (bf16_t)v.x; dst[1] = (bf16_t)v.y;
            dst[2] = (bf16_t)v.z; dst[3] = (bf16_t)v.w;
        } else {
            dst[0] = (bf16_t)0.f; dst[1] = (bf16_t)0.f;
            dst[2] = (bf16_t)0.f; dst[3] = (bf16_t)0.f;
        }
    }
    __syncthreads();

    int lane = tid & 31, w = tid >> 5;        // 8 waves
    int rlo = lane & 15, rsel = lane >> 4;

    // ========== GEMM1: h[64 x 1024] = x @ W1^T, +bias, relu ==========
    // wave w owns h columns [w*128, w*128+128), as four 32-col chunks;
    // per chunk: 4 m-tiles x 2 n-tiles = 8 accumulators (64 VGPRs)
    for (int nb = 0; nb < 4; ++nb) {
        v8f acc[4][2] = {};                   // [m-tile][n-tile]
        const bf16_t* Wp = w1b + ((size_t)e * HID + w * 128 + nb * 32) * DID;
        for (int k = 0; k < DID; k += 32) {
            v16bf a[4];
#pragma unroll
            for (int mt = 0; mt < 4; ++mt)
                a[mt] = ld_a(sx + mt * 16 * LDX + k, LDX, lane);
#pragma unroll
            for (int t = 0; t < 2; ++t) {
                v16bf bfr = ld_b(Wp + t * 16 * DID + k, DID, lane);
#pragma unroll
                for (int mt = 0; mt < 4; ++mt)
                    acc[mt][t] = WMMA_BF16(a[mt], bfr, acc[mt][t]);
            }
        }
#pragma unroll
        for (int t = 0; t < 2; ++t) {
            int col = w * 128 + nb * 32 + t * 16 + rlo;
            float bias = b1[e * HID + col];
#pragma unroll
            for (int mt = 0; mt < 4; ++mt) {
#pragma unroll
                for (int j = 0; j < 8; ++j) {
                    float v = acc[mt][t][j] + bias;
                    v = v > 0.f ? v : 0.f;    // relu
                    sh[(mt * 16 + j + (rsel << 3)) * LDH + col] = (bf16_t)v;
                }
            }
        }
    }
    __syncthreads();

    // ========== GEMM2: y[64 x 512] = h @ W2^T, +bias, *gate ==========
    // wave w owns y columns [w*64, w*64+64), as two 32-col chunks
    for (int nb = 0; nb < 2; ++nb) {
        v8f acc[4][2] = {};
        const bf16_t* W2p = w2b + ((size_t)e * DID + w * 64 + nb * 32) * HID;
        for (int k = 0; k < HID; k += 32) {
            v16bf a[4];
#pragma unroll
            for (int mt = 0; mt < 4; ++mt)
                a[mt] = ld_a(sh + mt * 16 * LDH + k, LDH, lane);
#pragma unroll
            for (int t = 0; t < 2; ++t) {
                v16bf bfr = ld_b(W2p + t * 16 * HID + k, HID, lane);
#pragma unroll
                for (int mt = 0; mt < 4; ++mt)
                    acc[mt][t] = WMMA_BF16(a[mt], bfr, acc[mt][t]);
            }
        }
#pragma unroll
        for (int t = 0; t < 2; ++t) {
            int col = w * 64 + nb * 32 + t * 16 + rlo;
            float bias = b2[e * DID + col];
#pragma unroll
            for (int mt = 0; mt < 4; ++mt) {
#pragma unroll
                for (int j = 0; j < 8; ++j) {
                    int m = mt * 16 + j + (rsel << 3);
                    int g = s_tok[m];
                    if (g >= 0)
                        out[(size_t)g * DID + col] = (acc[mt][t][j] + bias) * s_gate[m];
                }
            }
        }
    }
}

// ========================== launcher ====================================
extern "C" void kernel_launch(void* const* d_in, const int* in_sizes, int n_in,
                              void* d_out, int out_size, void* d_ws, size_t ws_size,
                              hipStream_t stream) {
    const float* inputs = (const float*)d_in[0];
    const float* embed  = (const float*)d_in[1];
    const float* rw     = (const float*)d_in[2];
    const float* w1     = (const float*)d_in[3];
    const float* b1     = (const float*)d_in[4];
    const float* w2     = (const float*)d_in[5];
    const float* b2     = (const float*)d_in[6];
    const unsigned char* mask = (const unsigned char*)d_in[7];
    float* out = (float*)d_out;

    char* ws = (char*)d_ws;
    int*    counts = (int*)ws;
    float*  gate   = (float*)(ws + WS_GATE);
    int*    perm   = (int*)(ws + WS_PERM);
    bf16_t* w1b    = (bf16_t*)(ws + WS_W1);
    bf16_t* w2b    = (bf16_t*)(ws + WS_W2);

    moe_prep<<<2048, 256, 0, stream>>>(w1, w2, w1b, w2b, counts);
    moe_router<<<NTOK / 8, 256, 0, stream>>>(inputs, embed, rw, mask, gate, perm, counts);
    // worst-case tile count: NTOK/TM + (NEXP-1) = 259
    moe_ffn<<<NTOK / TM + NEXP - 1, 256, 0, stream>>>(inputs, b1, b2, w1b, w2b,
                                                      gate, perm, counts, out);
}
